// BASE_14224931684905
// MI455X (gfx1250) — compile-verified, matrix operand order checked by reading, not run on hardware
//
#include <hip/hip_runtime.h>
#include <cstdint>
#include <cstddef>

typedef __attribute__((ext_vector_type(16))) _Float16 v16h;
typedef __attribute__((ext_vector_type(8)))  _Float16 v8h;
typedef __attribute__((ext_vector_type(8)))  float    v8f;

// ---------------------------------------------------------------------------
// WMMA GEMM:  C[M,N] = A[M,K] * Bt[N,K]^T (+ bias[m])
// A, Bt are f16 row-major. 128 threads (4 waves) cooperate on a 64x64 output
// tile. Per 32-deep K step, the block stages a 64x32 A tile and 64x32 B tile
// into LDS with GLOBAL_LOAD_ASYNC_TO_LDS_B128 (ASYNCcnt), double-buffered;
// each wave computes a 32x32 subtile as 2x2 v_wmma_f32_16x16x32_f16.
// LDS rows padded to 80B (5*16B, 5 coprime 16) so the 16-lane ds_load_b128
// row gather is bank-conflict free.
// Fragment layout per CDNA5 ISA (16-bit A 16x32): lane<16 -> K[0..7],K[16..23],
// lane>=16 -> K[8..15],K[24..31]; B mirrored with N on lanes.
// M,N multiples of 64; K multiple of 32.
// ---------------------------------------------------------------------------
#define LDSROW 40           // halfs per LDS row (32 data + 8 pad)
#define TILEB  (64 * LDSROW * 2)  // bytes per tile buffer (5120)

__global__ __launch_bounds__(128) void k_gemm_wmma(
    const _Float16* __restrict__ A, const _Float16* __restrict__ Bt,
    const float* __restrict__ bias, float* __restrict__ C,
    int M, int N, int K)
{
  __shared__ __align__(16) _Float16 smem[4][64 * LDSROW]; // A0,A1,B0,B1

  const int t    = threadIdx.x;
  const int lane = t & 31;
  const int w    = t >> 5;
  const int r15  = lane & 15;
  const int hi   = lane >> 4;
  const int kb   = hi << 3;                // 0 or 8
  const int tn   = blockIdx.x << 6;
  const int tm   = blockIdx.y << 6;
  const int wm   = (w >> 1) << 5;          // wave's 32x32 subtile
  const int wn   = (w & 1) << 5;

  const _Float16* Abase = A  + (size_t)tm * K;
  const _Float16* Bbase = Bt + (size_t)tn * K;

  // this thread's 4 async 16B copy slots: chunks 0..255 = A tile, 256..511 = B
  uint32_t ldsoff0[4];
  const _Float16* gsrc[4];
  #pragma unroll
  for (int j = 0; j < 4; ++j) {
    int c   = t + j * 128;
    int isB = c >> 8, cc = c & 255;
    int row = cc >> 2, q = cc & 3;
    gsrc[j]    = (isB ? Bbase : Abase) + (size_t)row * K + q * 8;
    ldsoff0[j] = (uint32_t)(uintptr_t)&smem[isB ? 2 : 0][row * LDSROW + q * 8];
  }

  auto issue_tile = [&](int k0, int buf) {
    #pragma unroll
    for (int j = 0; j < 4; ++j) {
      uint32_t lo = ldsoff0[j] + buf * TILEB;
      unsigned long long ga = (unsigned long long)(uintptr_t)(gsrc[j] + k0);
      asm volatile("global_load_async_to_lds_b128 %0, %1, off"
                   :: "v"(lo), "v"(ga) : "memory");
    }
  };

  v8f acc00 = {}, acc01 = {}, acc10 = {}, acc11 = {};
  const int nsteps = K >> 5;

  issue_tile(0, 0);
  for (int i = 0; i < nsteps; ++i) {
    const int cur = i & 1;
    asm volatile("s_wait_asynccnt 0x0" ::: "memory");  // own tile copies done
    __syncthreads();                                    // whole tile resident
    if (i + 1 < nsteps) issue_tile((i + 1) << 5, 1 - cur); // overlaps compute

    const _Float16* sA = smem[cur];
    const _Float16* sB = smem[2 + cur];
    v8h a0lo = *(const v8h*)(sA + (wm + r15) * LDSROW + kb);
    v8h a0hi = *(const v8h*)(sA + (wm + r15) * LDSROW + 16 + kb);
    v8h a1lo = *(const v8h*)(sA + (wm + 16 + r15) * LDSROW + kb);
    v8h a1hi = *(const v8h*)(sA + (wm + 16 + r15) * LDSROW + 16 + kb);
    v8h b0lo = *(const v8h*)(sB + (wn + r15) * LDSROW + kb);
    v8h b0hi = *(const v8h*)(sB + (wn + r15) * LDSROW + 16 + kb);
    v8h b1lo = *(const v8h*)(sB + (wn + 16 + r15) * LDSROW + kb);
    v8h b1hi = *(const v8h*)(sB + (wn + 16 + r15) * LDSROW + 16 + kb);

    v16h a0 = __builtin_shufflevector(a0lo, a0hi, 0,1,2,3,4,5,6,7,8,9,10,11,12,13,14,15);
    v16h a1 = __builtin_shufflevector(a1lo, a1hi, 0,1,2,3,4,5,6,7,8,9,10,11,12,13,14,15);
    v16h b0 = __builtin_shufflevector(b0lo, b0hi, 0,1,2,3,4,5,6,7,8,9,10,11,12,13,14,15);
    v16h b1 = __builtin_shufflevector(b1lo, b1hi, 0,1,2,3,4,5,6,7,8,9,10,11,12,13,14,15);

    acc00 = __builtin_amdgcn_wmma_f32_16x16x32_f16(false, a0, false, b0, (short)0, acc00, false, false);
    acc01 = __builtin_amdgcn_wmma_f32_16x16x32_f16(false, a0, false, b1, (short)0, acc01, false, false);
    acc10 = __builtin_amdgcn_wmma_f32_16x16x32_f16(false, a1, false, b0, (short)0, acc10, false, false);
    acc11 = __builtin_amdgcn_wmma_f32_16x16x32_f16(false, a1, false, b1, (short)0, acc11, false, false);
  }

  // C/D layout: lanes 0-15: (M=r, N=lane); lanes 16-31: (M=8+r, N=lane-16)
  #pragma unroll
  for (int r = 0; r < 8; ++r) {
    int mrow = (hi ? 8 + r : r);
    int m0 = tm + wm + mrow, m1 = tm + wm + 16 + mrow;
    int n0 = tn + wn + r15,  n1 = tn + wn + 16 + r15;
    float b0 = bias ? bias[m0] : 0.f;
    float b1 = bias ? bias[m1] : 0.f;
    C[(size_t)m0 * N + n0] = acc00[r] + b0;
    C[(size_t)m0 * N + n1] = acc01[r] + b0;
    C[(size_t)m1 * N + n0] = acc10[r] + b1;
    C[(size_t)m1 * N + n1] = acc11[r] + b1;
  }
}

// ---------------------------------------------------------------------------
// Conversion / layout kernels
// ---------------------------------------------------------------------------
__global__ void k_cvt(_Float16* __restrict__ dst, const float* __restrict__ src, int n) {
  int i = blockIdx.x * 256 + threadIdx.x;
  if (i < n) dst[i] = (_Float16)src[i];
}

// dst[n*K + k] = src[k*N + n]  (K x N f32 -> N x K f16)
__global__ void k_cvt_tr(_Float16* __restrict__ dst, const float* __restrict__ src, int K, int N) {
  int i = blockIdx.x * 256 + threadIdx.x;
  if (i >= K * N) return;
  int n = i / K, k = i - n * K;
  dst[i] = (_Float16)src[(size_t)k * N + n];
}

// concat-channel transpose for down/fuse GEMM B: dst[pos*1024+k]
__global__ void k_cat_tr(_Float16* __restrict__ dst, const float* __restrict__ a,
                         const float* __restrict__ b) {
  int i = blockIdx.x * 256 + threadIdx.x;
  if (i >= 1024 * 1024) return;
  int pos = i >> 10, k = i & 1023;
  float v = (k < 512) ? a[(size_t)k * 1024 + pos] : b[(size_t)(k - 512) * 1024 + pos];
  dst[i] = (_Float16)v;
}

// ---------------------------------------------------------------------------
// Squeeze-Excitation
// ---------------------------------------------------------------------------
__global__ __launch_bounds__(256) void k_chan_mean(const float* __restrict__ x, float* __restrict__ m) {
  __shared__ float sm[256];
  int c = blockIdx.x, t = threadIdx.x;
  const float* row = x + (size_t)c * 1024;
  float s = row[t] + row[t + 256] + row[t + 512] + row[t + 768];
  sm[t] = s; __syncthreads();
  for (int o = 128; o; o >>= 1) { if (t < o) sm[t] += sm[t + o]; __syncthreads(); }
  if (!t) m[c] = sm[0] * (1.f / 1024.f);
}

__global__ __launch_bounds__(512) void k_se_fc(const float* __restrict__ mvec,
    const float* __restrict__ w1, const float* __restrict__ b1,
    const float* __restrict__ w2, const float* __restrict__ b2,
    float* __restrict__ svec) {
  __shared__ float sm[512];
  __shared__ float h1[32];
  int t = threadIdx.x;
  sm[t] = mvec[t];
  __syncthreads();
  if (t < 32) {
    float acc = b1[t];
    for (int i = 0; i < 512; ++i) acc += w1[t * 512 + i] * sm[i];
    h1[t] = acc > 0.f ? acc : 0.f;
  }
  __syncthreads();
  float acc = b2[t];
  for (int i = 0; i < 32; ++i) acc += w2[t * 32 + i] * h1[i];
  svec[t] = 1.f / (1.f + __expf(-acc));
}

__global__ void k_scale(const float* __restrict__ x, const float* __restrict__ s,
                        float* __restrict__ dst) {
  int i = blockIdx.x * 256 + threadIdx.x;
  if (i < 512 * 1024) dst[i] = x[i] * s[i >> 10];
}

// ---------------------------------------------------------------------------
// gnconv pieces
// ---------------------------------------------------------------------------
__global__ void k_dwconv7(const float* __restrict__ fused, const float* __restrict__ w,
                          const float* __restrict__ b, float* __restrict__ dst) {
  int i = blockIdx.x * 256 + threadIdx.x;
  if (i >= 896 * 1024) return;
  int c = i >> 10, pos = i & 1023, y = pos >> 5, x = pos & 31;
  const float* src = fused + (size_t)(128 + c) * 1024;
  const float* wc  = w + c * 49;
  float acc = b[c];
  for (int dy = 0; dy < 7; ++dy) {
    int yy = y + dy - 3; if (yy < 0 || yy > 31) continue;
    for (int dx = 0; dx < 7; ++dx) {
      int xx = x + dx - 3; if (xx < 0 || xx > 31) continue;
      acc += src[yy * 32 + xx] * wc[dy * 7 + dx];
    }
  }
  dst[i] = acc;
}

__global__ void k_mul(float* __restrict__ dst, const float* __restrict__ a,
                      const float* __restrict__ b, int n) {
  int i = blockIdx.x * 256 + threadIdx.x;
  if (i < n) dst[i] = a[i] * b[i];
}

// ---------------------------------------------------------------------------
// RAL pieces
// ---------------------------------------------------------------------------
__global__ void k_bilinear16(const float* __restrict__ src, float* __restrict__ dst) {
  int i = blockIdx.x * 256 + threadIdx.x;
  if (i >= 512 * 256) return;
  int c = i >> 8, pos = i & 255, iy = pos >> 4, ix = pos & 15;
  float fy = iy * (31.f / 15.f), fx = ix * (31.f / 15.f);
  int y0 = (int)floorf(fy); int y1 = y0 + 1 > 31 ? 31 : y0 + 1; float wy = fy - (float)y0;
  int x0 = (int)floorf(fx); int x1 = x0 + 1 > 31 ? 31 : x0 + 1; float wx = fx - (float)x0;
  const float* s = src + (size_t)c * 1024;
  float r0 = s[y0 * 32 + x0] * (1.f - wx) + s[y0 * 32 + x1] * wx;
  float r1 = s[y1 * 32 + x0] * (1.f - wx) + s[y1 * 32 + x1] * wx;
  dst[i] = r0 * (1.f - wy) + r1 * wy;
}

// patch/im2col matrix P[p, c*9 + dy*3 + dx] over the 16x16 fgs (pad 1)
__global__ void k_patch3(const float* __restrict__ fgs, float* __restrict__ P) {
  int i = blockIdx.x * 256 + threadIdx.x;
  if (i >= 256 * 4608) return;
  int p = i / 4608, k = i - p * 4608;
  int c = k / 9, r = k - c * 9, dy = r / 3, dx = r - dy * 3;
  int py = p >> 4, px = p & 15;
  int yy = py + dy - 1, xx = px + dx - 1;
  float v = 0.f;
  if (yy >= 0 && yy < 16 && xx >= 0 && xx < 16) v = fgs[(size_t)c * 256 + yy * 16 + xx];
  P[i] = v;
}

__global__ __launch_bounds__(256) void k_rownorm(const float* __restrict__ P, float* __restrict__ nrm) {
  __shared__ float sm[256];
  int p = blockIdx.x, t = threadIdx.x;
  float s = 0.f;
  for (int k = t; k < 4608; k += 256) { float v = P[(size_t)p * 4608 + k]; s += v * v; }
  sm[t] = s; __syncthreads();
  for (int o = 128; o; o >>= 1) { if (t < o) sm[t] += sm[t + o]; __syncthreads(); }
  if (!t) nrm[p] = sqrtf(sm[0]);
}

__global__ void k_fgpn(const float* __restrict__ P, const float* __restrict__ nrm,
                       _Float16* __restrict__ dst) {
  int i = blockIdx.x * 256 + threadIdx.x;
  if (i >= 256 * 4608) return;
  int p = i / 4608;
  float d = nrm[p]; d = d > 1e-4f ? d : 1e-4f;
  dst[i] = (_Float16)(P[i] / d);
}

// column softmax of score*10 over p (256) per position t, stored transposed f16
__global__ __launch_bounds__(256) void k_softmax_attT(const float* __restrict__ score,
                                                      _Float16* __restrict__ attT) {
  __shared__ float sm[256];
  __shared__ float stat[2];
  int t = blockIdx.x, p = threadIdx.x;
  float v = score[(size_t)p * 256 + t] * 10.f;
  sm[p] = v; __syncthreads();
  for (int o = 128; o; o >>= 1) { if (p < o) sm[p] = fmaxf(sm[p], sm[p + o]); __syncthreads(); }
  if (!p) stat[0] = sm[0];
  __syncthreads();
  float e = __expf(v - stat[0]);
  sm[p] = e; __syncthreads();
  for (int o = 128; o; o >>= 1) { if (p < o) sm[p] += sm[p + o]; __syncthreads(); }
  if (!p) stat[1] = sm[0];
  __syncthreads();
  attT[(size_t)t * 256 + p] = (_Float16)(e / stat[1]);
}

// Bgp^T for paste GEMM: dst[n*256 + p], n = o*16 + m*4 + q, value bg[o, 2py-1+m, 2px-1+q]
__global__ void k_bgp(const float* __restrict__ bg, _Float16* __restrict__ dst) {
  int i = blockIdx.x * 256 + threadIdx.x;
  if (i >= 8192 * 256) return;
  int n = i >> 8, p = i & 255;
  int o = n >> 4, m = (n >> 2) & 3, q = n & 3;
  int py = p >> 4, px = p & 15;
  int Y = 2 * py - 1 + m, X = 2 * px - 1 + q;
  float v = 0.f;
  if (Y >= 0 && Y < 32 && X >= 0 && X < 32) v = bg[(size_t)o * 1024 + Y * 32 + X];
  dst[i] = (_Float16)v;
}

// gather overlapped patch paste: out[o,Y,X] = 0.25 * sum over <=4 (ty,tx) taps
__global__ void k_paste(const float* __restrict__ W, float* __restrict__ dst) {
  int i = blockIdx.x * 256 + threadIdx.x;
  if (i >= 512 * 1024) return;
  int o = i >> 10, pos = i & 1023, Y = pos >> 5, X = pos & 31;
  float acc = 0.f;
  int m0 = (Y + 1) & 1, n0 = (X + 1) & 1;
  for (int a = 0; a < 2; ++a) {
    int m = m0 + 2 * a, ty2 = Y + 1 - m;
    if (ty2 < 0 || ty2 > 30) continue;
    int ty = ty2 >> 1;
    for (int b = 0; b < 2; ++b) {
      int q = n0 + 2 * b, tx2 = X + 1 - q;
      if (tx2 < 0 || tx2 > 30) continue;
      int tx = tx2 >> 1;
      acc += W[(size_t)(ty * 16 + tx) * 8192 + o * 16 + m * 4 + q];
    }
  }
  dst[i] = acc * 0.25f;
}

// ---------------------------------------------------------------------------
// CSA: per-position 3x3 channel-dot softmax attention. Output row-major (p,c)
// so the flat buffer *is* the reshaped NCHW tensor.
// ---------------------------------------------------------------------------
__global__ __launch_bounds__(256) void k_csa(const float* __restrict__ act, float* __restrict__ outc) {
  __shared__ float sm[256];
  __shared__ float a9[9];
  __shared__ float att9[9];
  int pos = blockIdx.x, py = pos >> 5, px = pos & 31;
  int t = threadIdx.x;
  float sc0 = 1.f / (1.f + __expf(-act[(size_t)t * 1024 + pos]));
  float sc1 = 1.f / (1.f + __expf(-act[(size_t)(t + 256) * 1024 + pos]));
  for (int j = 0; j < 9; ++j) {
    int yy = py + j / 3 - 1, xx = px + j % 3 - 1;
    float part = 0.f;
    if (yy >= 0 && yy < 32 && xx >= 0 && xx < 32) {
      int nb = yy * 32 + xx;
      part = sc0 * (1.f / (1.f + __expf(-act[(size_t)t * 1024 + nb])))
           + sc1 * (1.f / (1.f + __expf(-act[(size_t)(t + 256) * 1024 + nb])));
    }
    sm[t] = part; __syncthreads();
    for (int o = 128; o; o >>= 1) { if (t < o) sm[t] += sm[t + o]; __syncthreads(); }
    if (!t) a9[j] = sm[0] * (1.f / 512.f);
    __syncthreads();
  }
  if (!t) {
    float mx = a9[0];
    for (int j = 1; j < 9; ++j) mx = fmaxf(mx, a9[j]);
    float s = 0.f;
    for (int j = 0; j < 9; ++j) { att9[j] = __expf(a9[j] - mx); s += att9[j]; }
    float inv = 1.f / s;
    for (int j = 0; j < 9; ++j) att9[j] *= inv;
  }
  __syncthreads();
  for (int cc = 0; cc < 2; ++cc) {
    int c = t + cc * 256;
    float acc = 0.f;
    for (int j = 0; j < 9; ++j) {
      int yy = py + j / 3 - 1, xx = px + j % 3 - 1;
      if (yy >= 0 && yy < 32 && xx >= 0 && xx < 32)
        acc += att9[j] * act[(size_t)c * 1024 + yy * 32 + xx];
    }
    outc[(size_t)pos * 512 + c] = acc;   // row-major (p, c) == reshaped NCHW
  }
}

// ---------------------------------------------------------------------------
// InstanceNorm + LeakyReLU(0.2), per channel row of 1024, in place
// ---------------------------------------------------------------------------
__global__ __launch_bounds__(256) void k_inorm_lrelu(float* __restrict__ buf) {
  __shared__ float s1[256], s2[256];
  int c = blockIdx.x, t = threadIdx.x;
  float* row = buf + (size_t)c * 1024;
  float vals[4]; float a = 0.f, b = 0.f;
  for (int j = 0; j < 4; ++j) { float v = row[t + j * 256]; vals[j] = v; a += v; b += v * v; }
  s1[t] = a; s2[t] = b; __syncthreads();
  for (int o = 128; o; o >>= 1) { if (t < o) { s1[t] += s1[t + o]; s2[t] += s2[t + o]; } __syncthreads(); }
  float mean = s1[0] * (1.f / 1024.f);
  float var  = s2[0] * (1.f / 1024.f) - mean * mean;
  float rs = rsqrtf(var + 1e-5f);
  for (int j = 0; j < 4; ++j) {
    float v = (vals[j] - mean) * rs;
    row[t + j * 256] = v > 0.f ? v : 0.2f * v;
  }
}

// ---------------------------------------------------------------------------
// Host orchestration
// ---------------------------------------------------------------------------
static inline int g1(int n) { return (n + 255) / 256; }

static void run_gnconv(const float* in, float* outp,
    const _Float16* hWpin, const float* pin_b,
    const float* dw_w, const float* dw_b,
    const _Float16* hWpw0, const float* pw0_b,
    const _Float16* hWpw1, const float* pw1_b,
    const _Float16* hWpout, const float* pout_b,
    _Float16* hB, float* f_fused, float* f_dw,
    float* f_t128, float* f_t256, float* f_t512, hipStream_t stream)
{
  k_cvt_tr<<<g1(512 * 1024), 256, 0, stream>>>(hB, in, 512, 1024);
  k_gemm_wmma<<<dim3(1024 / 64, 1024 / 64), 128, 0, stream>>>(hWpin, hB, pin_b, f_fused, 1024, 1024, 512);
  k_dwconv7<<<g1(896 * 1024), 256, 0, stream>>>(f_fused, dw_w, dw_b, f_dw);
  k_mul<<<g1(128 * 1024), 256, 0, stream>>>(f_t128, f_fused, f_dw, 128 * 1024);
  k_cvt_tr<<<g1(128 * 1024), 256, 0, stream>>>(hB, f_t128, 128, 1024);
  k_gemm_wmma<<<dim3(1024 / 64, 256 / 64), 128, 0, stream>>>(hWpw0, hB, pw0_b, f_t256, 256, 1024, 128);
  k_mul<<<g1(256 * 1024), 256, 0, stream>>>(f_t256, f_t256, f_dw + 128 * 1024, 256 * 1024);
  k_cvt_tr<<<g1(256 * 1024), 256, 0, stream>>>(hB, f_t256, 256, 1024);
  k_gemm_wmma<<<dim3(1024 / 64, 512 / 64), 128, 0, stream>>>(hWpw1, hB, pw1_b, f_t512, 512, 1024, 256);
  k_mul<<<g1(512 * 1024), 256, 0, stream>>>(f_t512, f_t512, f_dw + 384 * 1024, 512 * 1024);
  k_cvt_tr<<<g1(512 * 1024), 256, 0, stream>>>(hB, f_t512, 512, 1024);
  k_gemm_wmma<<<dim3(1024 / 64, 512 / 64), 128, 0, stream>>>(hWpout, hB, pout_b, outp, 512, 1024, 512);
}

extern "C" void kernel_launch(void* const* d_in, const int* in_sizes, int n_in,
                              void* d_out, int out_size, void* d_ws, size_t ws_size,
                              hipStream_t stream)
{
  (void)in_sizes; (void)n_in; (void)out_size; (void)ws_size;
  const float* x      = (const float*)d_in[0];
  const float* gus    = (const float*)d_in[1];
  const float* se_w1  = (const float*)d_in[2];
  const float* se_b1  = (const float*)d_in[3];
  const float* se_w2  = (const float*)d_in[4];
  const float* se_b2  = (const float*)d_in[5];
  const float* pin_w  = (const float*)d_in[6];
  const float* pin_b  = (const float*)d_in[7];
  const float* dw_w   = (const float*)d_in[8];
  const float* dw_b   = (const float*)d_in[9];
  const float* pw0_w  = (const float*)d_in[10];
  const float* pw0_b  = (const float*)d_in[11];
  const float* pw1_w  = (const float*)d_in[12];
  const float* pw1_b  = (const float*)d_in[13];
  const float* pout_w = (const float*)d_in[14];
  const float* pout_b = (const float*)d_in[15];
  const float* down_w = (const float*)d_in[16];
  const float* fuse_w = (const float*)d_in[17];
  float* out = (float*)d_out;

  char* base = (char*)d_ws; size_t off = 0;
  auto carve = [&](size_t bytes) -> void* {
    void* p = base + off; off += (bytes + 255) & ~(size_t)255; return p;
  };

  // f32 buffers
  float* f_out32 = (float*)carve(512 * 1024 * 4);   // SE output
  float* f_fused = (float*)carve(1024 * 1024 * 4);
  float* f_dw    = (float*)carve(896 * 1024 * 4);
  float* f_t128  = (float*)carve(128 * 1024 * 4);
  float* f_t256  = (float*)carve(256 * 1024 * 4);
  float* f_t512  = (float*)carve(512 * 1024 * 4);
  float* f_res   = (float*)carve(512 * 1024 * 4);   // out_res
  float* f_ral   = (float*)carve(512 * 1024 * 4);
  float* f_gn2   = (float*)carve(512 * 1024 * 4);
  float* f_fgs   = (float*)carve(512 * 256 * 4);
  float* f_P     = (float*)carve(256 * 4608 * 4);
  float* f_nrm   = (float*)carve(256 * 4);
  float* f_score = (float*)carve(256 * 256 * 4);
  float* f_W     = (float*)carve(256 * 8192 * 4);
  float* f_R     = (float*)carve(1024 * 512 * 4);   // gus result (p,c) row-major
  float* f_S     = (float*)carve(1024 * 512 * 4);   // csa result (p,c) row-major
  float* f_ydown = (float*)carve(512 * 1024 * 4);
  float* f_mvec  = (float*)carve(512 * 4);
  float* f_svec  = (float*)carve(512 * 4);
  // f16 staging
  _Float16* hA     = (_Float16*)carve((size_t)256 * 4608 * 2);  // >= 1024*1024
  _Float16* hB     = (_Float16*)carve((size_t)8192 * 256 * 2);  // >= 1024*1024
  _Float16* hWpin  = (_Float16*)carve((size_t)1024 * 512 * 2);
  _Float16* hWpw0  = (_Float16*)carve((size_t)256 * 128 * 2);
  _Float16* hWpw1  = (_Float16*)carve((size_t)512 * 256 * 2);
  _Float16* hWpout = (_Float16*)carve((size_t)512 * 512 * 2);
  _Float16* hWdown = (_Float16*)carve((size_t)512 * 1024 * 2);
  _Float16* hWfuse = (_Float16*)carve((size_t)512 * 1024 * 2);
  _Float16* hGus   = (_Float16*)carve((size_t)1024 * 1024 * 2);

  // ---- weight conversions (f32 -> f16 row-major) ----
  k_cvt<<<g1(1024 * 512), 256, 0, stream>>>(hWpin, pin_w, 1024 * 512);
  k_cvt<<<g1(256 * 128), 256, 0, stream>>>(hWpw0, pw0_w, 256 * 128);
  k_cvt<<<g1(512 * 256), 256, 0, stream>>>(hWpw1, pw1_w, 512 * 256);
  k_cvt<<<g1(512 * 512), 256, 0, stream>>>(hWpout, pout_w, 512 * 512);
  k_cvt<<<g1(512 * 1024), 256, 0, stream>>>(hWdown, down_w, 512 * 1024);
  k_cvt<<<g1(512 * 1024), 256, 0, stream>>>(hWfuse, fuse_w, 512 * 1024);
  k_cvt<<<g1(1024 * 1024), 256, 0, stream>>>(hGus, gus, 1024 * 1024);

  // ---- SE ----
  k_chan_mean<<<512, 256, 0, stream>>>(x, f_mvec);
  k_se_fc<<<1, 512, 0, stream>>>(f_mvec, se_w1, se_b1, se_w2, se_b2, f_svec);
  k_scale<<<g1(512 * 1024), 256, 0, stream>>>(x, f_svec, f_out32);

  // ---- out_res = gnconv(out32) ----
  run_gnconv(f_out32, f_res, hWpin, pin_b, dw_w, dw_b, hWpw0, pw0_b,
             hWpw1, pw1_b, hWpout, pout_b, hB, f_fused, f_dw,
             f_t128, f_t256, f_t512, stream);

  // ---- RAL(out32, out32) ----
  k_bilinear16<<<g1(512 * 256), 256, 0, stream>>>(f_out32, f_fgs);
  k_patch3<<<g1(256 * 4608), 256, 0, stream>>>(f_fgs, f_P);
  k_rownorm<<<256, 256, 0, stream>>>(f_P, f_nrm);
  k_fgpn<<<g1(256 * 4608), 256, 0, stream>>>(f_P, f_nrm, hA);     // A = fgp_n
  k_cvt<<<g1(256 * 4608), 256, 0, stream>>>(hB, f_P, 256 * 4608); // Bt = P
  k_gemm_wmma<<<dim3(256 / 64, 256 / 64), 128, 0, stream>>>(hA, hB, nullptr, f_score, 256, 256, 4608);
  k_softmax_attT<<<256, 256, 0, stream>>>(f_score, hA);           // A = Att^T
  k_bgp<<<g1(8192 * 256), 256, 0, stream>>>(f_out32, hB);         // Bt = Bgp^T
  k_gemm_wmma<<<dim3(8192 / 64, 256 / 64), 128, 0, stream>>>(hA, hB, nullptr, f_W, 256, 8192, 256);
  k_paste<<<g1(512 * 1024), 256, 0, stream>>>(f_W, f_ral);

  // ---- second gnconv ----
  run_gnconv(f_ral, f_gn2, hWpin, pin_b, dw_w, dw_b, hWpw0, pw0_b,
             hWpw1, pw1_b, hWpout, pout_b, hB, f_fused, f_dw,
             f_t128, f_t256, f_t512, stream);

  // ---- gus einsum: R[p,c] = gus[p,:] . gn2[c,:]  (Bt == gn2 layout) ----
  k_cvt<<<g1(512 * 1024), 256, 0, stream>>>(hB, f_gn2, 512 * 1024);
  k_gemm_wmma<<<dim3(512 / 64, 1024 / 64), 128, 0, stream>>>(hGus, hB, nullptr, f_R, 1024, 512, 1024);

  // ---- CSA ----
  k_csa<<<1024, 256, 0, stream>>>(f_gn2, f_S);

  // ---- down: concat(gus_out, csa) -> 1x1 conv -> inorm+lrelu ----
  k_cat_tr<<<g1(1024 * 1024), 256, 0, stream>>>(hB, f_R, f_S);
  k_gemm_wmma<<<dim3(1024 / 64, 512 / 64), 128, 0, stream>>>(hWdown, hB, nullptr, f_ydown, 512, 1024, 1024);
  k_inorm_lrelu<<<512, 256, 0, stream>>>(f_ydown);

  // ---- fuse: concat(ydown, out_res) -> 1x1 conv -> inorm+lrelu -> out ----
  k_cat_tr<<<g1(1024 * 1024), 256, 0, stream>>>(hB, f_ydown, f_res);
  k_gemm_wmma<<<dim3(1024 / 64, 512 / 64), 128, 0, stream>>>(hWfuse, hB, nullptr, out, 512, 1024, 1024);
  k_inorm_lrelu<<<512, 256, 0, stream>>>(out);
}
